// OPTAttentionInteger_42236708389055
// MI455X (gfx1250) — 8-bit matrix operand order repaired and checked by reading, compiled, not run on hardware
//
#include <hip/hip_runtime.h>
#include <hip/hip_bf16.h>


typedef __attribute__((ext_vector_type(8))) int v8i;

#define BATCH 2
#define SEQ   2048
#define DMODEL 1024
#define HEADS 16
#define HD    64
#define MROWS (BATCH * SEQ)

__device__ __forceinline__ int quant8i(float x) {
    int r = __float2int_rn(x * 32.0f);
    r = r < -128 ? -128 : r;
    r = r > 127 ? 127 : r;
    return r;
}

__device__ __forceinline__ v8i load_a8(const signed char* base, int lda) {
    int lane = threadIdx.x & 31;
    int row = lane & 15;
    int kb = (lane >> 4) << 4;
    const signed char* p = base + row * lda + kb;
    v8i a;
    a[0] = *(const int*)(p + 0);
    a[1] = *(const int*)(p + 4);
    a[2] = *(const int*)(p + 8);
    a[3] = *(const int*)(p + 12);
    a[4] = *(const int*)(p + 32);
    a[5] = *(const int*)(p + 36);
    a[6] = *(const int*)(p + 40);
    a[7] = *(const int*)(p + 44);
    return a;
}

__device__ __forceinline__ v8i load_b8(const signed char* base, int ldb) {
    int lane = threadIdx.x & 31;
    int n = lane & 15;
    int kb = (lane >> 4) << 4;
    const signed char* p = base + n * ldb + kb;
    v8i b;
    b[0] = *(const int*)(p + 0);
    b[1] = *(const int*)(p + 4);
    b[2] = *(const int*)(p + 8);
    b[3] = *(const int*)(p + 12);
    b[4] = *(const int*)(p + 32);
    b[5] = *(const int*)(p + 36);
    b[6] = *(const int*)(p + 40);
    b[7] = *(const int*)(p + 44);
    return b;
}

__global__ void __launch_bounds__(256) quantize_i8_kernel(const float* __restrict__ x,
                                                          signed char* __restrict__ y, int n) {
    int i = blockIdx.x * blockDim.x + threadIdx.x;
    if (i < n) y[i] = (signed char)quant8i(x[i]);
}

__global__ void __launch_bounds__(256) quantize_f_kernel(const float* __restrict__ x,
                                                         float* __restrict__ y, int n) {
    int i = blockIdx.x * blockDim.x + threadIdx.x;
    if (i < n) y[i] = (float)quant8i(x[i]) * (1.0f / 32.0f);
}

__global__ void __launch_bounds__(128) gemm_i8_kernel(
    const signed char* __restrict__ A,
    const signed char* __restrict__ W,
    const float* __restrict__ bq,
    float postscale, int mode,
    signed char* __restrict__ out8, float* __restrict__ outf)
{
    const int K = DMODEL;
    const int NT2 = DMODEL / 64;
    const int MT2 = MROWS / 32;
    int wid = blockIdx.x * (blockDim.x >> 5) + (threadIdx.x >> 5);
    int mt2 = wid / NT2;
    int nt2 = wid - mt2 * NT2;
    if (mt2 >= MT2) return;

    v8i acc[2][4];
#pragma unroll
    for (int i = 0; i < 2; ++i)
#pragma unroll
        for (int j = 0; j < 4; ++j)
            acc[i][j] = (v8i){0, 0, 0, 0, 0, 0, 0, 0};

    const signed char* abase = A + (size_t)mt2 * 32 * K;
    const signed char* bbase = W + (size_t)nt2 * 64 * K;
    for (int k0 = 0; k0 < K; k0 += 64) {
        v8i a0 = load_a8(abase + k0, K);
        v8i a1 = load_a8(abase + (size_t)16 * K + k0, K);
        v8i b0 = load_b8(bbase + k0, K);
        v8i b1 = load_b8(bbase + (size_t)16 * K + k0, K);
        v8i b2 = load_b8(bbase + (size_t)32 * K + k0, K);
        v8i b3 = load_b8(bbase + (size_t)48 * K + k0, K);
        acc[0][0] = __builtin_amdgcn_wmma_i32_16x16x64_iu8(true, a0, true, b0, acc[0][0], false, false);
        acc[0][1] = __builtin_amdgcn_wmma_i32_16x16x64_iu8(true, a0, true, b1, acc[0][1], false, false);
        acc[0][2] = __builtin_amdgcn_wmma_i32_16x16x64_iu8(true, a0, true, b2, acc[0][2], false, false);
        acc[0][3] = __builtin_amdgcn_wmma_i32_16x16x64_iu8(true, a0, true, b3, acc[0][3], false, false);
        acc[1][0] = __builtin_amdgcn_wmma_i32_16x16x64_iu8(true, a1, true, b0, acc[1][0], false, false);
        acc[1][1] = __builtin_amdgcn_wmma_i32_16x16x64_iu8(true, a1, true, b1, acc[1][1], false, false);
        acc[1][2] = __builtin_amdgcn_wmma_i32_16x16x64_iu8(true, a1, true, b2, acc[1][2], false, false);
        acc[1][3] = __builtin_amdgcn_wmma_i32_16x16x64_iu8(true, a1, true, b3, acc[1][3], false, false);
    }

    int lane = threadIdx.x & 31;
    int n = lane & 15, hi = lane >> 4;
#pragma unroll
    for (int j = 0; j < 4; ++j) {
        int col = nt2 * 64 + j * 16 + n;
        float bb = bq[col];
        int h = col >> 6, hd = col & (HD - 1);
#pragma unroll
        for (int i = 0; i < 2; ++i) {
#pragma unroll
            for (int r = 0; r < 8; ++r) {
                int m = mt2 * 32 + i * 16 + r + 8 * hi;
                float f = ((float)acc[i][j][r]) * (1.0f / 1024.0f) + bb;
                f *= postscale;
                if (mode == 2) {
                    outf[(size_t)m * DMODEL + col] = f;
                } else {
                    int b_ = m >> 11, nseq = m & (SEQ - 1);
                    int bh = b_ * HEADS + h;
                    size_t idx = (mode == 0)
                                     ? ((size_t)bh * SEQ + nseq) * HD + hd
                                     : ((size_t)bh * HD + hd) * SEQ + nseq;
                    out8[idx] = (signed char)quant8i(f);
                }
            }
        }
    }
}

__global__ void __launch_bounds__(128) attn_i8_kernel(
    const signed char* __restrict__ Qq,
    const signed char* __restrict__ Kq,
    const signed char* __restrict__ Vt,
    const float* __restrict__ mask,
    signed char* __restrict__ Cq)
{
    __shared__ __align__(16) signed char probs[4][16 * 64];

    int wv = threadIdx.x >> 5;
    int lane = threadIdx.x & 31;
    int wid = blockIdx.x * 4 + wv;
    int bh = wid >> 7;
    int qt = wid & 127;
    int m0 = qt * 16;
    int b_ = bh >> 4, h = bh & (HEADS - 1);

    const signed char* qbase = Qq + ((size_t)bh * SEQ + m0) * HD;
    const signed char* kbase = Kq + (size_t)bh * SEQ * HD;
    const float* mbase = mask + (size_t)b_ * SEQ * SEQ;

    v8i aq = load_a8(qbase, HD);
    int n = lane & 15, hi = lane >> 4;
    const float inv1024 = 1.0f / 1024.0f;

    float vmax[8];
#pragma unroll
    for (int r = 0; r < 8; ++r) vmax[r] = -3.4e38f;
    for (int j0 = 0; j0 < SEQ; j0 += 16) {
        v8i bk = load_b8(kbase + (size_t)j0 * HD, HD);
        v8i c = {0, 0, 0, 0, 0, 0, 0, 0};
        c = __builtin_amdgcn_wmma_i32_16x16x64_iu8(true, aq, true, bk, c, false, false);
#pragma unroll
        for (int r = 0; r < 8; ++r) {
            int i = m0 + r + 8 * hi;
            float s = (float)c[r] * inv1024 + mbase[(size_t)i * SEQ + (j0 + n)];
            vmax[r] = fmaxf(vmax[r], s);
        }
    }
#pragma unroll
    for (int r = 0; r < 8; ++r)
        for (int off = 1; off < 16; off <<= 1)
            vmax[r] = fmaxf(vmax[r], __shfl_xor(vmax[r], off, 32));

    float vsum[8];
#pragma unroll
    for (int r = 0; r < 8; ++r) vsum[r] = 0.0f;
    for (int j0 = 0; j0 < SEQ; j0 += 16) {
        v8i bk = load_b8(kbase + (size_t)j0 * HD, HD);
        v8i c = {0, 0, 0, 0, 0, 0, 0, 0};
        c = __builtin_amdgcn_wmma_i32_16x16x64_iu8(true, aq, true, bk, c, false, false);
#pragma unroll
        for (int r = 0; r < 8; ++r) {
            int i = m0 + r + 8 * hi;
            float s = (float)c[r] * inv1024 + mbase[(size_t)i * SEQ + (j0 + n)];
            vsum[r] += __expf(s - vmax[r]);
        }
    }
    float vinv[8];
#pragma unroll
    for (int r = 0; r < 8; ++r) {
        for (int off = 1; off < 16; off <<= 1)
            vsum[r] += __shfl_xor(vsum[r], off, 32);
        vinv[r] = 1.0f / vsum[r];
    }

    v8i acc0 = {0,0,0,0,0,0,0,0}, acc1 = {0,0,0,0,0,0,0,0};
    v8i acc2 = {0,0,0,0,0,0,0,0}, acc3 = {0,0,0,0,0,0,0,0};
    for (int j0 = 0; j0 < SEQ; j0 += 64) {
#pragma unroll
        for (int jj = 0; jj < 4; ++jj) {
            v8i bk = load_b8(kbase + (size_t)(j0 + jj * 16) * HD, HD);
            v8i c = {0, 0, 0, 0, 0, 0, 0, 0};
            c = __builtin_amdgcn_wmma_i32_16x16x64_iu8(true, aq, true, bk, c, false, false);
#pragma unroll
            for (int r = 0; r < 8; ++r) {
                int i = m0 + r + 8 * hi;
                float s = (float)c[r] * inv1024 + mbase[(size_t)i * SEQ + (j0 + jj * 16 + n)];
                float p = __expf(s - vmax[r]) * vinv[r];
                probs[wv][(r + 8 * hi) * 64 + jj * 16 + n] = (signed char)quant8i(p);
            }
        }
        __syncthreads();
        v8i ap = load_a8(&probs[wv][0], 64);
        {
            const signed char* vb = Vt + ((size_t)bh * HD + 0 * 16) * SEQ + j0;
            acc0 = __builtin_amdgcn_wmma_i32_16x16x64_iu8(true, ap, true, load_b8(vb, SEQ), acc0, false, false);
            vb = Vt + ((size_t)bh * HD + 1 * 16) * SEQ + j0;
            acc1 = __builtin_amdgcn_wmma_i32_16x16x64_iu8(true, ap, true, load_b8(vb, SEQ), acc1, false, false);
            vb = Vt + ((size_t)bh * HD + 2 * 16) * SEQ + j0;
            acc2 = __builtin_amdgcn_wmma_i32_16x16x64_iu8(true, ap, true, load_b8(vb, SEQ), acc2, false, false);
            vb = Vt + ((size_t)bh * HD + 3 * 16) * SEQ + j0;
            acc3 = __builtin_amdgcn_wmma_i32_16x16x64_iu8(true, ap, true, load_b8(vb, SEQ), acc3, false, false);
        }
        __syncthreads();
    }

    v8i accs[4] = {acc0, acc1, acc2, acc3};
#pragma unroll
    for (int ht = 0; ht < 4; ++ht) {
#pragma unroll
        for (int r = 0; r < 8; ++r) {
            int seq = m0 + r + 8 * hi;
            int hd = ht * 16 + n;
            float f = (float)accs[ht][r] * inv1024;
            Cq[((size_t)(b_ * SEQ + seq)) * DMODEL + h * HD + hd] = (signed char)quant8i(f);
        }
    }
}

extern "C" void kernel_launch(void* const* d_in, const int* in_sizes, int n_in,
                              void* d_out, int out_size, void* d_ws, size_t ws_size,
                              hipStream_t stream) {
    (void)in_sizes; (void)n_in; (void)out_size; (void)ws_size;

    const float* hidden = (const float*)d_in[0];
    const float* amask  = (const float*)d_in[1];
    const float* q_w = (const float*)d_in[2]; const float* q_b = (const float*)d_in[3];
    const float* k_w = (const float*)d_in[4]; const float* k_b = (const float*)d_in[5];
    const float* v_w = (const float*)d_in[6]; const float* v_b = (const float*)d_in[7];
    const float* o_w = (const float*)d_in[8]; const float* o_b = (const float*)d_in[9];
    float* out = (float*)d_out;

    char* ws = (char*)d_ws;
    size_t off = 0;
    auto carve = [&](size_t bytes) -> char* {
        char* p = ws + off;
        off = (off + bytes + 255) & ~(size_t)255;
        return p;
    };
    signed char* Xq  = (signed char*)carve((size_t)MROWS * DMODEL);
    signed char* Wqq = (signed char*)carve((size_t)DMODEL * DMODEL);
    signed char* Wqk = (signed char*)carve((size_t)DMODEL * DMODEL);
    signed char* Wqv = (signed char*)carve((size_t)DMODEL * DMODEL);
    signed char* Wqo = (signed char*)carve((size_t)DMODEL * DMODEL);
    float* bfq = (float*)carve(DMODEL * sizeof(float));
    float* bfk = (float*)carve(DMODEL * sizeof(float));
    float* bfv = (float*)carve(DMODEL * sizeof(float));
    float* bfo = (float*)carve(DMODEL * sizeof(float));
    signed char* Qq = (signed char*)carve((size_t)BATCH * HEADS * SEQ * HD);
    signed char* Kq = (signed char*)carve((size_t)BATCH * HEADS * SEQ * HD);
    signed char* Vt = (signed char*)carve((size_t)BATCH * HEADS * HD * SEQ);
    signed char* Cq = (signed char*)carve((size_t)MROWS * DMODEL);

    {
        int n = MROWS * DMODEL;
        quantize_i8_kernel<<<(n + 255) / 256, 256, 0, stream>>>(hidden, Xq, n);
        n = DMODEL * DMODEL;
        quantize_i8_kernel<<<(n + 255) / 256, 256, 0, stream>>>(q_w, Wqq, n);
        quantize_i8_kernel<<<(n + 255) / 256, 256, 0, stream>>>(k_w, Wqk, n);
        quantize_i8_kernel<<<(n + 255) / 256, 256, 0, stream>>>(v_w, Wqv, n);
        quantize_i8_kernel<<<(n + 255) / 256, 256, 0, stream>>>(o_w, Wqo, n);
        n = DMODEL;
        quantize_f_kernel<<<(n + 255) / 256, 256, 0, stream>>>(q_b, bfq, n);
        quantize_f_kernel<<<(n + 255) / 256, 256, 0, stream>>>(k_b, bfk, n);
        quantize_f_kernel<<<(n + 255) / 256, 256, 0, stream>>>(v_b, bfv, n);
        quantize_f_kernel<<<(n + 255) / 256, 256, 0, stream>>>(o_b, bfo, n);
    }

    const int gemm_blocks = (MROWS / 32) * (DMODEL / 64) / 4;
    const float SCALING = 0.125f;
    gemm_i8_kernel<<<gemm_blocks, 128, 0, stream>>>(Xq, Wqq, bfq, SCALING, 0, Qq, nullptr);
    gemm_i8_kernel<<<gemm_blocks, 128, 0, stream>>>(Xq, Wqk, bfk, 1.0f,    0, Kq, nullptr);
    gemm_i8_kernel<<<gemm_blocks, 128, 0, stream>>>(Xq, Wqv, bfv, 1.0f,    1, Vt, nullptr);

    const int attn_blocks = (BATCH * HEADS) * (SEQ / 16) / 4;
    attn_i8_kernel<<<attn_blocks, 128, 0, stream>>>(Qq, Kq, Vt, amask, Cq);

    gemm_i8_kernel<<<gemm_blocks, 128, 0, stream>>>(Cq, Wqo, bfo, 1.0f, 2, nullptr, out);
}
